// Model3_4355096838495
// MI455X (gfx1250) — matrix operand order repaired and checked by reading, not compiled, not run on hardware
//
#include <hip/hip_runtime.h>

// ---------------------------------------------------------------------------
// Types matching CDNA5 WMMA operand shapes (wave32)
// ---------------------------------------------------------------------------
typedef __attribute__((ext_vector_type(2)))  float   v2f;    // f32 A/B frag (16x16x4)
typedef __attribute__((ext_vector_type(8)))  float   v8f;    // f32 C/D frag (16x16)
typedef __attribute__((ext_vector_type(8)))  __bf16  v8bf;   // half of a bf16 frag (16B)
typedef __attribute__((ext_vector_type(16))) __bf16  v16bf;  // bf16 A/B frag (16x32 / 32x16)

union BFrag {
    v16bf v;
    v8bf  h[2];
};

__device__ __forceinline__ v8f wmma_bf16(const BFrag& a, const BFrag& b, v8f c)
{
    return __builtin_amdgcn_wmma_f32_16x16x32_bf16(
        false, a.v, false, b.v, (short)0, c, false, false);
}

// ---------------------------------------------------------------------------
// CDNA5 async global->LDS copy (VGLOBAL, tracked on ASYNCcnt).
// LDS operand is the low 32 bits of the generic pointer: the LDS aperture maps
// addr[31:0] directly to the LDS offset (ISA 10.2).
// ---------------------------------------------------------------------------
__device__ __forceinline__ void async_g2l_b128(void* lds_ptr, const void* gptr)
{
    asm volatile("global_load_async_to_lds_b128 %0, %1, off"
                 :: "v"((unsigned int)(uintptr_t)lds_ptr), "v"(gptr)
                 : "memory");
}
__device__ __forceinline__ void wait_async0()
{
    asm volatile("s_wait_asynccnt 0x0" ::: "memory");
}
__device__ __forceinline__ void wait_async6()   // allow 6 newest in flight
{
    asm volatile("s_wait_asynccnt 0x6" ::: "memory");
}

// ---------------------------------------------------------------------------
// Kernel 1: T = relu(X @ W + b), exact fp32 via V_WMMA_F32_16X16X4_F32.
// Double-buffered LDS tiles fed by async-to-LDS; K-chunk = 32.
//   Block: 256 threads = 8 waves; block tile 128(M) x 64(N); wave tile 16x64.
// Also emits the bf16 hi/lo split of T for the bf16x3 score GEMM.
// ---------------------------------------------------------------------------
#define KT 32
__global__ __launch_bounds__(256) void transform_relu_f32(
    const float* __restrict__ X, const float* __restrict__ W,
    const float* __restrict__ bias,
    float* __restrict__ outF32, __bf16* __restrict__ outHi, __bf16* __restrict__ outLo,
    int M, int K, int N)
{
    __shared__ __attribute__((aligned(16))) float sX[2][128][36];
    __shared__ __attribute__((aligned(16))) float sW[2][KT][64];

    const int tid  = threadIdx.x;
    const int lane = tid & 31;
    const int wave = tid >> 5;
    const int m0   = blockIdx.x * 128;
    const int n0   = blockIdx.y * 64;

    auto stage = [&](int buf, int k0) {
#pragma unroll
        for (int c = 0; c < 4; ++c) {                      // X tile: 1024 chunks
            const int id = tid + 256 * c;
            const int r = id >> 3, q = id & 7;
            async_g2l_b128(&sX[buf][r][q * 4],
                           X + (size_t)(m0 + r) * K + k0 + q * 4);
        }
#pragma unroll
        for (int c = 0; c < 2; ++c) {                      // W tile: 512 chunks
            const int id = tid + 256 * c;
            const int r = id >> 4, q = id & 15;
            async_g2l_b128(&sW[buf][r][q * 4],
                           W + (size_t)(k0 + r) * N + n0 + q * 4);
        }
    };

    const int mrow = lane & 15;
    const int ncol = lane & 15;
    const int kb   = (lane >> 4) * 2;          // k sub-offset for f32 frags

    v8f acc[4] = {v8f{}, v8f{}, v8f{}, v8f{}};

    const int nchunks = K / KT;
    stage(0, 0);

    for (int ch = 0; ch < nchunks; ++ch) {
        const int buf = ch & 1;
        if (ch + 1 < nchunks) {
            stage(buf ^ 1, (ch + 1) * KT);     // prefetch next chunk
            wait_async6();                     // current chunk's 6 are done
        } else {
            wait_async0();
        }
        __syncthreads();

        const float* xrow = &sX[buf][wave * 16 + mrow][kb];
        const float* wcol = &sW[buf][0][0] + kb * 64 + ncol;
#pragma unroll
        for (int kk = 0; kk < KT; kk += 4) {
            v2f a = *(const v2f*)(xrow + kk);
#pragma unroll
            for (int f = 0; f < 4; ++f) {
                v2f b;
                b.x = wcol[kk * 64 + f * 16];
                b.y = wcol[kk * 64 + 64 + f * 16];
                acc[f] = __builtin_amdgcn_wmma_f32_16x16x4_f32(
                    false, a, false, b, (short)0, acc[f], false, false);
            }
        }
        __syncthreads();
    }

    const int rbase = (lane >> 4) * 8;
#pragma unroll
    for (int f = 0; f < 4; ++f) {
        const int   col = n0 + f * 16 + ncol;
        const float bv  = bias[col];
#pragma unroll
        for (int r = 0; r < 8; ++r) {
            float x = acc[f][r] + bv;
            x = fmaxf(x, 0.0f);
            const size_t off = (size_t)(m0 + rbase + r) * N + col;
            if (outF32) outF32[off] = x;
            __bf16 h = (__bf16)x;
            outHi[off] = h;
            outLo[off] = (__bf16)(x - (float)h);
        }
    }
}

// ---------------------------------------------------------------------------
// Kernel 2: blocked score GEMM (bf16x3) + per-block top-2 candidates.
//   Block tile 128(m: comments) x 128(n: srt rows), K=1024 in chunks of 32,
//   both operands async-staged to LDS. 8 waves in a 4(m)x2(n) grid; each wave
//   computes 32x64 (8 accumulators, 24 v_wmma per K-chunk).
//   LDS rows padded to 56 bf16 (112B): 16B-aligned for b128 async and a
//   28-DWORD stride (gcd(28,64)=4) is conflict-free for 16-lane frag reads.
//   Epilogue: dump 128x128 f32 scores to LDS (reusing the tile space), then
//   128 threads scan one row each for a per-block top-2 candidate pair.
// ---------------------------------------------------------------------------
#define SROW 56   // LDS row stride in bf16 elements (112 bytes)
__global__ __launch_bounds__(256) void scores_gemm_top2(
    const __bf16* __restrict__ TcHi, const __bf16* __restrict__ TcLo,
    const __bf16* __restrict__ TsHi, const __bf16* __restrict__ TsLo,
    float* __restrict__ candV, int* __restrict__ candI,
    int D /*1024*/, int NB /*64 n-blocks*/)
{
    __shared__ __attribute__((aligned(16))) char smem[65536];
    __bf16* sAhi = (__bf16*)(smem);             // 128 x SROW  (14336 B)
    __bf16* sAlo = (__bf16*)(smem + 14336);
    __bf16* sBhi = (__bf16*)(smem + 28672);
    __bf16* sBlo = (__bf16*)(smem + 43008);     // ends at 57344
    float*  dump = (float*)smem;                // 128x128 f32, reused after GEMM

    const int tid  = threadIdx.x;
    const int lane = tid & 31;
    const int wave = tid >> 5;
    const int wm   = wave & 3;                  // 4 m-groups of 32 rows
    const int wn   = wave >> 2;                 // 2 n-groups of 64 cols
    const int l15  = lane & 15;
    const int half = lane >> 4;

    const int m0 = blockIdx.y * 128;            // comment rows
    const int n0 = blockIdx.x * 128;            // srt rows (score columns)

    v8f acc[2][4] = {};

    for (int k0 = 0; k0 < D; k0 += 32) {
        // Stage A-hi/lo and B-hi/lo chunks: 4 x 512 b128 -> 8 asyncs/thread.
#pragma unroll
        for (int c = 0; c < 2; ++c) {
            const int id = tid + 256 * c;
            const int r = id >> 2, q = id & 3;
            const size_t gA = (size_t)(m0 + r) * D + k0 + q * 8;
            const size_t gB = (size_t)(n0 + r) * D + k0 + q * 8;
            const int loff = r * SROW + q * 8;
            async_g2l_b128(sAhi + loff, TcHi + gA);
            async_g2l_b128(sAlo + loff, TcLo + gA);
            async_g2l_b128(sBhi + loff, TsHi + gB);
            async_g2l_b128(sBlo + loff, TsLo + gB);
        }
        wait_async0();
        __syncthreads();

        BFrag ahi[2], alo[2];
#pragma unroll
        for (int t = 0; t < 2; ++t) {
            const int ar = (wm * 32 + t * 16 + l15) * SROW + half * 8;
            ahi[t].h[0] = *(const v8bf*)&sAhi[ar];
            ahi[t].h[1] = *(const v8bf*)&sAhi[ar + 16];
            alo[t].h[0] = *(const v8bf*)&sAlo[ar];
            alo[t].h[1] = *(const v8bf*)&sAlo[ar + 16];
        }
#pragma unroll
        for (int j = 0; j < 4; ++j) {
            const int br = (wn * 64 + j * 16 + l15) * SROW + half * 8;
            BFrag bhi, blo;
            bhi.h[0] = *(const v8bf*)&sBhi[br];
            bhi.h[1] = *(const v8bf*)&sBhi[br + 16];
            blo.h[0] = *(const v8bf*)&sBlo[br];
            blo.h[1] = *(const v8bf*)&sBlo[br + 16];
#pragma unroll
            for (int t = 0; t < 2; ++t) {
                acc[t][j] = wmma_bf16(ahi[t], bhi, acc[t][j]);
                acc[t][j] = wmma_bf16(ahi[t], blo, acc[t][j]);
                acc[t][j] = wmma_bf16(alo[t], bhi, acc[t][j]);
            }
        }
        __syncthreads();                         // tiles free for next chunk
    }

    // Dump the 128x128 f32 score tile into LDS (tile buffers are dead now).
#pragma unroll
    for (int t = 0; t < 2; ++t)
#pragma unroll
        for (int j = 0; j < 4; ++j)
#pragma unroll
            for (int r = 0; r < 8; ++r)
                dump[(wm * 32 + t * 16 + half * 8 + r) * 128 +
                     (wn * 64 + j * 16 + l15)] = acc[t][j][r];
    __syncthreads();

    // Per-block top-2 per row (ascending column order -> lowest index on ties).
    if (tid < 128) {
        const int row = tid;
        float b1 = -3.0e38f, b2 = -3.0e38f;
        int   i1 = 0, i2 = 0;
        for (int c = 0; c < 128; ++c) {
            const float v = dump[row * 128 + c];
            const int   i = n0 + c;
            if (v > b1)      { b2 = b1; i2 = i1; b1 = v; i1 = i; }
            else if (v > b2) { b2 = v;  i2 = i; }
        }
        const size_t base = ((size_t)(m0 + row) * NB + blockIdx.x) * 2;
        candV[base + 0] = b1;  candI[base + 0] = i1;
        candV[base + 1] = b2;  candI[base + 1] = i2;
    }
}

// ---------------------------------------------------------------------------
// Kernel 2b: fold 64 blocks x 2 candidates per row into the final top-2.
// ---------------------------------------------------------------------------
__global__ __launch_bounds__(256) void top2_reduce(
    const float* __restrict__ candV, const int* __restrict__ candI,
    int* __restrict__ topIdx, int NB)
{
    const int row = blockIdx.x * 256 + threadIdx.x;
    float b1 = -3.0e38f, b2 = -3.0e38f;
    int   i1 = 0, i2 = 0;
    const size_t base = (size_t)row * NB * 2;
    for (int c = 0; c < NB * 2; ++c) {
        const float v = candV[base + c];
        const int   i = candI[base + c];
        if (v > b1)      { b2 = b1; i2 = i1; b1 = v; i1 = i; }
        else if (v > b2) { b2 = v;  i2 = i; }
    }
    topIdx[row * 2 + 0] = i1;
    topIdx[row * 2 + 1] = i2;
}

// ---------------------------------------------------------------------------
// Kernel 3: gather selected fp32 Ts rows into the output [B, 2, D].
// ---------------------------------------------------------------------------
__global__ __launch_bounds__(256) void gather_rows(
    const float* __restrict__ TsF32, const int* __restrict__ topIdx,
    float* __restrict__ out, int D)
{
    const int     pair = blockIdx.x;               // 0 .. B*2-1
    const int     idx  = topIdx[pair];
    const float4* src  = (const float4*)(TsF32 + (size_t)idx * D);
    float4*       dst  = (float4*)(out + (size_t)pair * D);
    dst[threadIdx.x] = src[threadIdx.x];           // D/4 == 256 == blockDim.x
}

// ---------------------------------------------------------------------------
// Launcher
// ---------------------------------------------------------------------------
extern "C" void kernel_launch(void* const* d_in, const int* in_sizes, int n_in,
                              void* d_out, int out_size, void* d_ws, size_t ws_size,
                              hipStream_t stream)
{
    const float* srt = (const float*)d_in[0];   // [8192,1024]
    const float* com = (const float*)d_in[1];   // [4096, 768]
    const float* Wc  = (const float*)d_in[2];   // [ 768,1024]
    const float* bc  = (const float*)d_in[3];   // [1024]
    const float* Ws  = (const float*)d_in[4];   // [1024,1024]
    const float* bs  = (const float*)d_in[5];   // [1024]
    float*       out = (float*)d_out;           // [4096,2,1024]

    const int B = 4096, N = 8192, DC = 768, DS = 1024;
    const int NB = N / 128;                      // 64 n-blocks

    // Workspace carve-out (~86 MB).
    char*   w      = (char*)d_ws;
    float*  TsF32  = (float*)w;  w += (size_t)N * DS * sizeof(float);
    __bf16* TsHi   = (__bf16*)w; w += (size_t)N * DS * sizeof(__bf16);
    __bf16* TsLo   = (__bf16*)w; w += (size_t)N * DS * sizeof(__bf16);
    __bf16* TcHi   = (__bf16*)w; w += (size_t)B * DS * sizeof(__bf16);
    __bf16* TcLo   = (__bf16*)w; w += (size_t)B * DS * sizeof(__bf16);
    float*  candV  = (float*)w;  w += (size_t)B * NB * 2 * sizeof(float);
    int*    candI  = (int*)w;    w += (size_t)B * NB * 2 * sizeof(int);
    int*    topIdx = (int*)w;

    const dim3 blk(256);

    // Ts = relu(srt @ W_srt + b_srt): fp32 WMMA, keep fp32 + bf16 hi/lo.
    transform_relu_f32<<<dim3(N / 128, DS / 64), blk, 0, stream>>>(
        srt, Ws, bs, TsF32, TsHi, TsLo, N, DS, DS);

    // Tc = relu(comments @ W_comment + b_comment): bf16 hi/lo only.
    transform_relu_f32<<<dim3(B / 128, DS / 64), blk, 0, stream>>>(
        com, Wc, bc, nullptr, TcHi, TcLo, B, DC, DS);

    // Blocked bf16x3 score GEMM with per-block top-2 candidates.
    scores_gemm_top2<<<dim3(NB, B / 128), blk, 0, stream>>>(
        TcHi, TcLo, TsHi, TsLo, candV, candI, DS, NB);

    // Fold candidates into final top-2 indices (softmax is order-preserving).
    top2_reduce<<<dim3(B / 256), blk, 0, stream>>>(candV, candI, topIdx, NB);

    // Gather selected fp32 rows (k == 2, per the reference constant).
    gather_rows<<<dim3(B * 2), blk, 0, stream>>>(TsF32, topIdx, out, DS);
}
